// UniGCNRegression_89412629168657
// MI455X (gfx1250) — matrix operand order repaired and checked by reading, not compiled, not run on hardware
//
#include <hip/hip_runtime.h>
#include <hip/hip_bf16.h>

typedef __attribute__((ext_vector_type(16))) __bf16 v16bf;
typedef __attribute__((ext_vector_type(8)))  __bf16 v8bf;
typedef __attribute__((ext_vector_type(4)))  __bf16 v4bf;
typedef __attribute__((ext_vector_type(8)))  float  v8f;
typedef __attribute__((ext_vector_type(4)))  float  v4f;

#define H 128

// ---------------- degree / norm ----------------
__global__ __launch_bounds__(256) void deg_init_kernel(float* __restrict__ deg, int n) {
    int i = blockIdx.x * blockDim.x + threadIdx.x;
    if (i < n) deg[i] = 1.0f;  // self loop
}

__global__ __launch_bounds__(256) void deg_acc_kernel(const long long* __restrict__ dst,
                                                      float* __restrict__ deg, long long E) {
    long long e = (long long)blockIdx.x * blockDim.x + threadIdx.x;
    if (e < E) atomicAdd(&deg[(int)dst[e]], 1.0f);
}

__global__ __launch_bounds__(256) void dinv_kernel(const float* __restrict__ deg,
                                                   float* __restrict__ dinv, int n) {
    int i = blockIdx.x * blockDim.x + threadIdx.x;
    if (i < n) dinv[i] = rsqrtf(deg[i]);  // deg >= 1 always (self loops)
}

__global__ __launch_bounds__(256) void enorm_kernel(const long long* __restrict__ src,
                                                    const long long* __restrict__ dst,
                                                    const float* __restrict__ dinv,
                                                    float* __restrict__ enorm, long long E) {
    long long e = (long long)blockIdx.x * blockDim.x + threadIdx.x;
    if (e < E) enorm[e] = dinv[(int)src[e]] * dinv[(int)dst[e]];
}

// ---------------- layer 1: x(N x 6) @ W1(6 x 128) ----------------
__global__ __launch_bounds__(256) void gemm6_kernel(const float* __restrict__ x,
                                                    const float* __restrict__ W1,
                                                    float* __restrict__ out, int nrows) {
    int idx = blockIdx.x * blockDim.x + threadIdx.x;
    if (idx >= nrows * H) return;
    int n = idx >> 7, c = idx & (H - 1);
    const float* xr = x + (size_t)n * 6;
    float s = 0.f;
#pragma unroll
    for (int k = 0; k < 6; ++k) s += xr[k] * W1[k * H + c];
    out[idx] = s;
}

// ---------------- pack f32 weight (K x ncols) into WMMA bf16 B lane layout ----------------
// chunk = nt*Kt + kb ; within chunk: 32 lanes x 16 bf16.
// lane<16 : col = n0+lane, K = kb*32 + j (j=0..15)
// lane>=16: col = n0+lane-16, K = kb*32 + 16 + j
__global__ __launch_bounds__(256) void pack_b_kernel(const float* __restrict__ W,
                                                     __bf16* __restrict__ Bp,
                                                     int K, int ncols) {
    int wave = (blockIdx.x * blockDim.x + threadIdx.x) >> 5;
    int lane = threadIdx.x & 31;
    int Kt = K >> 5, Nt = ncols >> 4;
    if (wave >= Kt * Nt) return;
    int nt = wave / Kt, kb = wave % Kt;
    int half = lane >> 4, lm = lane & 15;
    int col = (nt << 4) + lm;
    __bf16* d = Bp + ((size_t)wave * 32 + lane) * 16;
#pragma unroll
    for (int j = 0; j < 16; ++j) {
        int k = (kb << 5) + (half << 4) + j;
        d[j] = (__bf16)W[(size_t)k * ncols + col];
    }
}

// ---------------- WMMA GEMM: C(N x NT*16) = A(N x 128, bf16) @ B, f32 acc ----------------
template <int NT>
__global__ __launch_bounds__(256) void wmma_gemm_kernel(const __bf16* __restrict__ A,
                                                        const __bf16* __restrict__ Bp,
                                                        float* __restrict__ C,
                                                        const float* __restrict__ bias,
                                                        int nrows, int do_relu) {
    constexpr int KT = 4;          // K = 128 = 4 * 32
    constexpr int NCOL = NT * 16;
    extern __shared__ char smem[];
    __bf16* ldsB = (__bf16*)smem;  // NT*16 cols x 128 K, chunked

    int tid = threadIdx.x;
    // cooperative stage of packed B into LDS (NT*4096 bytes)
    for (int i = tid; i < NT * 256; i += 256)
        ((float4*)ldsB)[i] = ((const float4*)Bp)[i];
    __syncthreads();

    int wave = tid >> 5;
    int lane = tid & 31;
    int half = lane >> 4, lm = lane & 15;
    int mtiles = (nrows + 15) >> 4;
    int tile = blockIdx.x * 8 + wave;
    if (tile >= mtiles) return;

    int row = (tile << 4) + lm;
    int rowc = row < nrows ? row : (nrows - 1);
    const __bf16* arow = A + (size_t)rowc * H;

    v8f acc[NT] = {};

#pragma unroll
    for (int kb = 0; kb < KT; ++kb) {
        int k0 = (kb << 5) + (half << 4);
        v8bf lo = *(const v8bf*)(arow + k0);        // K = k0 .. k0+7
        v8bf hi = *(const v8bf*)(arow + k0 + 8);    // K = k0+8 .. k0+15
        v16bf a = __builtin_shufflevector(lo, hi,
                                          0, 1, 2, 3, 4, 5, 6, 7,
                                          8, 9, 10, 11, 12, 13, 14, 15);
#pragma unroll
        for (int nt = 0; nt < NT; ++nt) {
            v16bf b = ((const v16bf*)ldsB)[(nt * KT + kb) * 32 + lane];
            acc[nt] = __builtin_amdgcn_wmma_f32_16x16x32_bf16(
                false, a, false, b, (short)0, acc[nt], false, false);
        }
    }

#pragma unroll
    for (int nt = 0; nt < NT; ++nt) {
        int col = (nt << 4) + lm;
        float bv = bias ? bias[col] : 0.f;
#pragma unroll
        for (int r = 0; r < 8; ++r) {
            int rr = (tile << 4) + r + (half << 3);
            if (rr < nrows) {
                float v = acc[nt][r] + bv;
                if (do_relu) v = v > 0.f ? v : 0.f;
                C[(size_t)rr * NCOL + col] = v;
            }
        }
    }
}

// ---------------- self-loop init: agg[i] = t[i] * dinv[i]^2 ----------------
__global__ __launch_bounds__(256) void selfloop_kernel(const float* __restrict__ t,
                                                       const float* __restrict__ dinv,
                                                       float* __restrict__ agg, int nrows) {
    int idx = blockIdx.x * blockDim.x + threadIdx.x;
    int n = idx >> 5, lane = idx & 31;
    if (n >= nrows) return;
    float w = dinv[n] * dinv[n];
    v4f v = *(const v4f*)(t + (size_t)n * H + lane * 4);
    *(v4f*)(agg + (size_t)n * H + lane * 4) = v * w;
}

// ---------------- edge scatter-add: one wave per edge ----------------
__global__ __launch_bounds__(256) void scatter_kernel(const float* __restrict__ t,
                                                      const long long* __restrict__ src,
                                                      const long long* __restrict__ dst,
                                                      const float* __restrict__ enorm,
                                                      float* __restrict__ agg, long long E) {
    long long e = ((long long)blockIdx.x * blockDim.x + threadIdx.x) >> 5;
    int lane = threadIdx.x & 31;
    if (e >= E) return;
    int s = (int)src[e], d = (int)dst[e];
    float w = enorm[e];
    v4f v = *(const v4f*)(t + (size_t)s * H + lane * 4);
    float* ap = agg + (size_t)d * H + lane * 4;
    atomicAdd(ap + 0, v[0] * w);
    atomicAdd(ap + 1, v[1] * w);
    atomicAdd(ap + 2, v[2] * w);
    atomicAdd(ap + 3, v[3] * w);
}

// ---------------- bias + LayerNorm (+ optional ReLU): one wave per node ----------------
// Output written in bf16: it is consumed only as the A operand of the next WMMA GEMM.
__global__ __launch_bounds__(256) void ln_kernel(const float* __restrict__ agg,
                                                 const float* __restrict__ bias,
                                                 const float* __restrict__ g,
                                                 const float* __restrict__ beta,
                                                 __bf16* __restrict__ out,
                                                 int nrows, int do_relu) {
    int gidx = blockIdx.x * blockDim.x + threadIdx.x;
    int n = gidx >> 5, lane = gidx & 31;
    if (n >= nrows) return;
    v4f v = *(const v4f*)(agg + (size_t)n * H + lane * 4);
    v4f bb = *(const v4f*)(bias + lane * 4);
    v += bb;
    float s = v[0] + v[1] + v[2] + v[3];
#pragma unroll
    for (int off = 16; off > 0; off >>= 1) s += __shfl_xor(s, off, 32);
    float mu = s * (1.0f / H);
    v4f d = v - mu;
    float q = d[0] * d[0] + d[1] * d[1] + d[2] * d[2] + d[3] * d[3];
#pragma unroll
    for (int off = 16; off > 0; off >>= 1) q += __shfl_xor(q, off, 32);
    float rstd = rsqrtf(q * (1.0f / H) + 1e-5f);
    v4f gg = *(const v4f*)(g + lane * 4);
    v4f be = *(const v4f*)(beta + lane * 4);
    v4bf o;
#pragma unroll
    for (int j = 0; j < 4; ++j) {
        float y = d[j] * rstd * gg[j] + be[j];
        if (do_relu) y = y > 0.f ? y : 0.f;
        o[j] = (__bf16)y;
    }
    *(v4bf*)(out + (size_t)n * H + lane * 4) = o;
}

// ---------------- head: sigmoid(h2(N x 64) @ rW2 + rb2), one wave per node ----------------
__global__ __launch_bounds__(256) void head2_kernel(const float* __restrict__ h2,
                                                    const float* __restrict__ rW2,
                                                    const float* __restrict__ rb2,
                                                    float* __restrict__ out, int nrows) {
    int gidx = blockIdx.x * blockDim.x + threadIdx.x;
    int n = gidx >> 5, lane = gidx & 31;
    if (n >= nrows) return;
    float2 v = *(const float2*)(h2 + (size_t)n * 64 + lane * 2);
    float s = v.x * rW2[lane * 2] + v.y * rW2[lane * 2 + 1];
#pragma unroll
    for (int off = 16; off > 0; off >>= 1) s += __shfl_xor(s, off, 32);
    if (lane == 0) out[n] = 1.0f / (1.0f + __expf(-(s + rb2[0])));
}

// ---------------- host ----------------
extern "C" void kernel_launch(void* const* d_in, const int* in_sizes, int n_in,
                              void* d_out, int out_size, void* d_ws, size_t ws_size,
                              hipStream_t stream) {
    const float*     x    = (const float*)d_in[0];
    const long long* ei   = (const long long*)d_in[1];
    const float* W1  = (const float*)d_in[2];  const float* b1  = (const float*)d_in[3];
    const float* W2  = (const float*)d_in[4];  const float* b2  = (const float*)d_in[5];
    const float* W3  = (const float*)d_in[6];  const float* b3  = (const float*)d_in[7];
    const float* g1  = (const float*)d_in[8];  const float* be1 = (const float*)d_in[9];
    const float* g2  = (const float*)d_in[10]; const float* be2 = (const float*)d_in[11];
    const float* g3  = (const float*)d_in[12]; const float* be3 = (const float*)d_in[13];
    const float* rW1 = (const float*)d_in[14]; const float* rb1 = (const float*)d_in[15];
    const float* rW2 = (const float*)d_in[16]; const float* rb2 = (const float*)d_in[17];
    float* out = (float*)d_out;

    const long long N = in_sizes[0] / 6;
    const long long E = in_sizes[1] / 2;
    const long long* srcIdx = ei;
    const long long* dstIdx = ei + E;

    float*  wf     = (float*)d_ws;
    float*  deg    = wf;
    float*  dinv   = deg + N;
    float*  enorm  = dinv + N;
    float*  bufT   = enorm + E;          // GEMM output f32 / also holds h2 (N x 64) at head
    float*  bufAgg = bufT + N * H;
    __bf16* bufH   = (__bf16*)(bufAgg + N * H);   // LN output, bf16 (N x 128)
    __bf16* W2p    = bufH + N * H;
    __bf16* W3p    = W2p + H * H;
    __bf16* rW1p   = W3p + H * H;

    const int T = 256;
    dim3 blk(T);
    auto cdiv = [](long long a, long long b) { return (unsigned)((a + b - 1) / b); };

    // degrees / norms
    deg_init_kernel<<<cdiv(N, T), blk, 0, stream>>>(deg, (int)N);
    deg_acc_kernel<<<cdiv(E, T), blk, 0, stream>>>(dstIdx, deg, E);
    dinv_kernel<<<cdiv(N, T), blk, 0, stream>>>(deg, dinv, (int)N);
    enorm_kernel<<<cdiv(E, T), blk, 0, stream>>>(srcIdx, dstIdx, dinv, enorm, E);

    // pack weights once into WMMA B layout
    pack_b_kernel<<<cdiv(32 * 32, T), blk, 0, stream>>>(W2, W2p, H, H);   // 32 waves
    pack_b_kernel<<<cdiv(32 * 32, T), blk, 0, stream>>>(W3, W3p, H, H);
    pack_b_kernel<<<cdiv(16 * 32, T), blk, 0, stream>>>(rW1, rW1p, H, 64);

    const unsigned gNodeWave = cdiv(N * 32, T);
    const unsigned gEdgeWave = cdiv(E * 32, T);
    const unsigned gGemm     = cdiv((N + 15) / 16, 8);

    // layer 1
    gemm6_kernel<<<cdiv(N * H, T), blk, 0, stream>>>(x, W1, bufT, (int)N);
    selfloop_kernel<<<gNodeWave, blk, 0, stream>>>(bufT, dinv, bufAgg, (int)N);
    scatter_kernel<<<gEdgeWave, blk, 0, stream>>>(bufT, srcIdx, dstIdx, enorm, bufAgg, E);
    ln_kernel<<<gNodeWave, blk, 0, stream>>>(bufAgg, b1, g1, be1, bufH, (int)N, 1);

    // layer 2
    wmma_gemm_kernel<8><<<gGemm, blk, 8 * 4096, stream>>>(bufH, W2p, bufT, nullptr, (int)N, 0);
    selfloop_kernel<<<gNodeWave, blk, 0, stream>>>(bufT, dinv, bufAgg, (int)N);
    scatter_kernel<<<gEdgeWave, blk, 0, stream>>>(bufT, srcIdx, dstIdx, enorm, bufAgg, E);
    ln_kernel<<<gNodeWave, blk, 0, stream>>>(bufAgg, b2, g2, be2, bufH, (int)N, 1);

    // layer 3 (no relu)
    wmma_gemm_kernel<8><<<gGemm, blk, 8 * 4096, stream>>>(bufH, W3p, bufT, nullptr, (int)N, 0);
    selfloop_kernel<<<gNodeWave, blk, 0, stream>>>(bufT, dinv, bufAgg, (int)N);
    scatter_kernel<<<gEdgeWave, blk, 0, stream>>>(bufT, srcIdx, dstIdx, enorm, bufAgg, E);
    ln_kernel<<<gNodeWave, blk, 0, stream>>>(bufAgg, b3, g3, be3, bufH, (int)N, 0);

    // head: h2 = relu(h @ rW1 + rb1) -> bufT (N x 64), then sigmoid dot
    wmma_gemm_kernel<4><<<gGemm, blk, 4 * 4096, stream>>>(bufH, rW1p, bufT, rb1, (int)N, 1);
    head2_kernel<<<gNodeWave, blk, 0, stream>>>(bufT, rW2, rb2, out, (int)N);
}
